// STGN_41695542510361
// MI455X (gfx1250) — matrix operand order repaired and checked by reading, not compiled, not run on hardware
//
#include <hip/hip_runtime.h>
#include <cmath>

// ---------------------------------------------------------------------------
// STGN fused recurrent kernel for MI455X (gfx1250, wave32, WMMA)
//   gates[k] = x[s] @ W[k]^T  (+ h @ U[k]^T for k<4), all 8 gates kept in
//   v8f WMMA accumulators per wave; elementwise cell update fused in-register.
// ---------------------------------------------------------------------------

typedef __attribute__((ext_vector_type(16))) __bf16 v16bf;
typedef __attribute__((ext_vector_type(8)))  float  v8f;

#define S_STEPS 128
#define UL_DIM  1024
#define H_DIM   256

// Trivial aligned POD types (HIP's uint4 has ctors -> not union-safe)
struct U4 { unsigned int x, y, z, w; } __attribute__((aligned(16)));
struct US4 { unsigned short x, y, z, w; } __attribute__((aligned(8)));
struct F4 { float x, y, z, w; } __attribute__((aligned(16)));

union BfFrag { v16bf v; U4 q[2]; };

__device__ __forceinline__ unsigned short f32_to_bf16_rne(float f) {
  union { float f; unsigned int u; } cv; cv.f = f;
  unsigned int u = cv.u;
  return (unsigned short)((u + 0x7FFFu + ((u >> 16) & 1u)) >> 16);
}

__device__ __forceinline__ float sigmoidf_(float x) {
  return 1.0f / (1.0f + __expf(-x));
}

// ---------------------------------------------------------------------------
// Pack W (gates 0..7) and U (gates 8..11) into bf16, B-fragment-ready layout:
//   Bpk[((g*8 + kc)*16 + ntg)*512 + lane*16 + e]
//     = src_g[j*H + kc*32 + 16*(lane/16) + e],  j = ntg*16 + lane%16
// so a wave's B fragment (32x16 bf16, K chunk kc, N tile ntg) is 32 contiguous
// bytes per lane -> two global_load_b128 per lane in the step kernel.
// ---------------------------------------------------------------------------
__global__ void __launch_bounds__(256)
stgn_pack_weights(const float* __restrict__ W, const float* __restrict__ U,
                  unsigned short* __restrict__ Bpk) {
  int tid = blockIdx.x * blockDim.x + threadIdx.x;   // 0 .. 12*8*16*32-1
  int g     = tid >> 12;
  int rem   = tid & 4095;
  int kc    = rem >> 9;
  int rem2  = rem & 511;
  int ntg   = rem2 >> 5;
  int lane  = rem2 & 31;
  int khalf = lane >> 4;
  int n     = lane & 15;
  int j     = ntg * 16 + n;
  const float* src = (g < 8) ? (W + (size_t)g * H_DIM * H_DIM)
                             : (U + (size_t)(g - 8) * H_DIM * H_DIM);
  const float* p = src + (size_t)j * H_DIM + kc * 32 + khalf * 16;
  union { unsigned short h16[16]; U4 q[2]; } pk;
#pragma unroll
  for (int e = 0; e < 16; ++e) pk.h16[e] = f32_to_bf16_rne(p[e]);
  U4* dst = (U4*)(Bpk + (size_t)tid * 16);
  dst[0] = pk.q[0];
  dst[1] = pk.q[1];
}

// ---------------------------------------------------------------------------
// One GEMM phase: 8 K-chunks of 32 over a 32-row A block (f32 -> bf16 staged
// in padded LDS), NG gates of B fragments from the packed weights.
// ---------------------------------------------------------------------------
template<int GBASE, int NG>
__device__ __forceinline__ void gemm_phase(
    const float* __restrict__ Asrc,              // 32-row block, row stride H
    const unsigned short* __restrict__ Bpk,
    unsigned short* Abuf,                        // [32][40] bf16 (padded)
    v8f* acc, int tid, int lane, int mt, int ntg)
{
  const int arow  = tid >> 3;      // 0..31
  const int acg   = tid & 7;       // col group of 4 floats
  const int khalf = lane >> 4;
  const int nlo   = lane & 15;
#pragma unroll 1
  for (int kc = 0; kc < 8; ++kc) {
    __syncthreads();
    // stage 32x32 f32 tile -> bf16 LDS (coalesced b128 loads, b64 LDS stores)
    F4 v4 = *(const F4*)(Asrc + (size_t)arow * H_DIM + kc * 32 + acg * 4);
    US4 h4;
    h4.x = f32_to_bf16_rne(v4.x);
    h4.y = f32_to_bf16_rne(v4.y);
    h4.z = f32_to_bf16_rne(v4.z);
    h4.w = f32_to_bf16_rne(v4.w);
    *(US4*)(&Abuf[arow * 40 + acg * 4]) = h4;
    if (kc < 7)
      __builtin_prefetch(Asrc + (size_t)arow * H_DIM + (kc + 1) * 32 + acg * 4, 0, 1);
    __syncthreads();
    // A fragment (16x32 bf16): two ds_load_b128 per lane
    BfFrag af;
    const unsigned short* ar = &Abuf[(mt * 16 + nlo) * 40 + khalf * 16];
    af.q[0] = *(const U4*)(ar);        // K = 16*khalf + 0..7
    af.q[1] = *(const U4*)(ar + 8);    // K = 16*khalf + 8..15
#pragma unroll
    for (int gi = 0; gi < NG; ++gi) {
      const int g = GBASE + gi;
      const unsigned short* bp =
          Bpk + (((size_t)((g * 8 + kc) * 16 + ntg)) << 9) + lane * 16;
      BfFrag bf_;
      bf_.q[0] = ((const U4*)bp)[0];
      bf_.q[1] = ((const U4*)bp)[1];
      acc[gi] = __builtin_amdgcn_wmma_f32_16x16x32_bf16(
          false, af.v, false, bf_.v, (short)0, acc[gi], false, false);
    }
  }
}

// ---------------------------------------------------------------------------
// One recurrent step: workgroup = 8 waves = 32(M) x 64(N) output tile.
// Wave (mt,nt) owns one 16x16 tile for ALL 8 gates -> fused elementwise.
// ---------------------------------------------------------------------------
__global__ void __launch_bounds__(256)
stgn_step(const float* __restrict__ x,    // (S,UL,H)
          const float* __restrict__ dtv,  // (S,UL)
          const float* __restrict__ dsv,  // (S,UL)
          const float* __restrict__ Hin,  // (UL,H) = h_{s-1}
          const float* __restrict__ Cin,  // (UL,H) = c_{s-1}
          const unsigned short* __restrict__ Bpk,
          const float* __restrict__ b,    // (8,H)
          const float* __restrict__ V,    // (6,H)
          float* __restrict__ out,        // (S,UL,H)
          float* __restrict__ cbuf,       // (UL,H)
          float* __restrict__ hlast, float* __restrict__ clast,
          int s, int is_last)
{
  __shared__ unsigned short Abuf[32 * 40];   // padded row stride: 40 halves
  const int tid   = threadIdx.x;
  const int lane  = tid & 31;
  const int w     = tid >> 5;
  const int mt    = w & 1;
  const int nt    = w >> 1;                 // 0..3
  const int khalf = lane >> 4;
  const int nlo   = lane & 15;
  const int u0    = blockIdx.x * 32;
  const int n0    = blockIdx.y * 64;
  const int ntg   = (n0 >> 4) + nt;

  v8f acc[8] = {};

  // gates 0..7 += x[s] @ W[k]^T
  gemm_phase<0, 8>(x + ((size_t)s * UL_DIM + u0) * H_DIM, Bpk, Abuf, acc,
                   tid, lane, mt, ntg);
  // gates 0..3 += h @ U[k]^T   (packed as gates 8..11)
  gemm_phase<8, 4>(Hin + (size_t)u0 * H_DIM, Bpk, Abuf, acc,
                   tid, lane, mt, ntg);

  // -------- fused elementwise cell update (C-layout: M=r+8*khalf, N=nlo) ----
  const int j = n0 + nt * 16 + nlo;
  const float b0 = b[0 * H_DIM + j], b1 = b[1 * H_DIM + j];
  const float b2 = b[2 * H_DIM + j], b3 = b[3 * H_DIM + j];
  const float b4 = b[4 * H_DIM + j], b5 = b[5 * H_DIM + j];
  const float b6 = b[6 * H_DIM + j], b7 = b[7 * H_DIM + j];
  const float vTt1 = V[0 * H_DIM + j], vTt2 = V[1 * H_DIM + j];
  const float vTo  = V[2 * H_DIM + j], vDd1 = V[3 * H_DIM + j];
  const float vDd2 = V[4 * H_DIM + j], vDo  = V[5 * H_DIM + j];

#pragma unroll
  for (int r = 0; r < 8; ++r) {
    const int u = u0 + mt * 16 + r + 8 * khalf;
    const float tt    = dtv[(size_t)s * UL_DIM + u];
    const float st    = dsv[(size_t)s * UL_DIM + u];
    const float cprev = Cin[(size_t)u * H_DIM + j];
    const float gi_ = sigmoidf_(acc[0][r] + b0);
    const float gf  = sigmoidf_(acc[1][r] + b1);
    const float go  = sigmoidf_(acc[2][r] + tt * vTo + st * vDo + b2);
    const float ct  = tanhf(acc[3][r] + b3);
    const float T1  = sigmoidf_(acc[4][r] + sigmoidf_(tt * vTt1) + b4);
    const float D1  = sigmoidf_(acc[5][r] + sigmoidf_(st * vDd1) + b5);
    const float T2  = sigmoidf_(acc[6][r] + sigmoidf_(tt * vTt2) + b6);
    const float D2  = sigmoidf_(acc[7][r] + sigmoidf_(st * vDd2) + b7);
    const float fc   = gf * cprev;
    const float ict  = gi_ * ct;
    const float chat = fc + ict * T1 * D1;
    const float cnew = fc + ict * T2 * D2;
    const float hnew = go * tanhf(chat);
    out[((size_t)s * UL_DIM + u) * H_DIM + j] = hnew;
    cbuf[(size_t)u * H_DIM + j] = cnew;
    if (is_last) {
      hlast[(size_t)u * H_DIM + j] = hnew;
      clast[(size_t)u * H_DIM + j] = cnew;
    }
  }
}

// ---------------------------------------------------------------------------
extern "C" void kernel_launch(void* const* d_in, const int* in_sizes, int n_in,
                              void* d_out, int out_size, void* d_ws, size_t ws_size,
                              hipStream_t stream) {
  (void)in_sizes; (void)n_in; (void)out_size; (void)ws_size;
  const float* x   = (const float*)d_in[0];
  const float* dtv = (const float*)d_in[1];
  const float* dsv = (const float*)d_in[2];
  const float* h0  = (const float*)d_in[3];
  const float* c0  = (const float*)d_in[4];
  const float* W   = (const float*)d_in[5];
  const float* U   = (const float*)d_in[6];
  const float* V   = (const float*)d_in[7];
  const float* b   = (const float*)d_in[8];
  float* out = (float*)d_out;

  unsigned short* Bpk = (unsigned short*)d_ws;                 // 1.5 MB packed bf16 weights
  float* cbuf = (float*)((char*)d_ws + (2u << 20));            // 1 MB c state

  stgn_pack_weights<<<192, 256, 0, stream>>>(W, U, Bpk);

  float* hlast = out + (size_t)S_STEPS * UL_DIM * H_DIM;
  float* clast = hlast + (size_t)UL_DIM * H_DIM;

  dim3 grid(UL_DIM / 32, H_DIM / 64);   // 32 x 4 workgroups, 256 threads each
  for (int s = 0; s < S_STEPS; ++s) {
    const float* Hin = (s == 0) ? h0 : (out + (size_t)(s - 1) * UL_DIM * H_DIM);
    const float* Cin = (s == 0) ? c0 : cbuf;
    stgn_step<<<grid, 256, 0, stream>>>(x, dtv, dsv, Hin, Cin, Bpk, b, V,
                                        out, cbuf, hlast, clast, s,
                                        (s == S_STEPS - 1) ? 1 : 0);
  }
}
